// GARNN_45372034515229
// MI455X (gfx1250) — matrix operand order repaired and checked by reading, not compiled, not run on hardware
//
#include <hip/hip_runtime.h>
#include <hip/hip_bf16.h>

#define B_  32
#define T_  12
#define N_  1024
#define D_  64
#define L_  2
#define D3_ 192

#define AS1 __attribute__((address_space(1)))
#define AS3 __attribute__((address_space(3)))

#if defined(__has_builtin)
#  if __has_builtin(__builtin_amdgcn_global_load_async_to_lds_b128) && \
      __has_builtin(__builtin_amdgcn_s_wait_asynccnt)
#    define USE_ASYNC_LDS 1
#  endif
#endif
#ifndef USE_ASYNC_LDS
#  define USE_ASYNC_LDS 0
#endif

typedef __attribute__((ext_vector_type(16))) __bf16 v16bf;
typedef __attribute__((ext_vector_type(8)))  __bf16 v8bf;
typedef __attribute__((ext_vector_type(8)))  float  v8f;
typedef __attribute__((ext_vector_type(4)))  int    v4i;

typedef unsigned short u16;
typedef unsigned int   u32;

__device__ __forceinline__ float bf2f(u16 u) {
    return (float)__builtin_bit_cast(__bf16, u);
}
__device__ __forceinline__ u16 f2bf(float f) {
    return __builtin_bit_cast(u16, (__bf16)f);   // native cvt, RNE
}
__device__ __forceinline__ v8f vzero() {
    v8f z = {0.f,0.f,0.f,0.f,0.f,0.f,0.f,0.f};
    return z;
}
__device__ __forceinline__ v8f wmma_bf16(v16bf a, v16bf b, v8f c) {
    // D = A(16x32 bf16) * B(32x16 bf16) + C(16x16 f32)
    return __builtin_amdgcn_wmma_f32_16x16x32_bf16(false, a, false, b, (short)0, c, false, false);
}
// A-operand (16x32): lane<16 holds row=lane, K = 0..15; lane>=16 K = 16..31.
// Caller passes pointer to first 8-half chunk (row*ld + 32*kt + 16*h); 2nd chunk is +8 halfs.
__device__ __forceinline__ v16bf load_a16(const u16* p0) {
    v8bf lo = *(const v8bf*)p0;
    v8bf hi = *(const v8bf*)(p0 + 8);
    v16bf r;
#pragma unroll
    for (int i = 0; i < 8; ++i) { r[i] = lo[i]; r[i + 8] = hi[i]; }
    return r;
}
// B-operand (32x16): lane<16 holds col=lane, K=0..15; lane>=16 K=16..31 -> 16 contiguous halfs.
__device__ __forceinline__ v16bf load_b16(const u16* p) {
    return *(const v16bf*)p;
}

// ---------------- weight convert (fp32 [L,rows,cols] -> bf16 transposed [L,cols,rows]) --------
__global__ void wconv_kernel(const float* __restrict__ src, u16* __restrict__ dst,
                             int rows, int cols) {
    int per = rows * cols;
    int idx = blockIdx.x * blockDim.x + threadIdx.x;
    if (idx >= L_ * per) return;
    int l = idx / per, rem = idx % per;
    int r = rem / cols, c = rem % cols;
    dst[(size_t)l * per + (size_t)c * rows + r] = f2bf(src[(size_t)l * per + (size_t)r * cols + c]);
}

// ---------------- fp32 -> bf16 (row-major + transposed) for z and hidden[:,l] -----------------
__global__ void tobf16_kernel(const float* __restrict__ zsrc, long long zstride,
                              const float* __restrict__ hsrc, long long hstride,
                              u16* __restrict__ zbf, u16* __restrict__ zT,
                              u16* __restrict__ hbf, u16* __restrict__ hT) {
    int idx = blockIdx.x * blockDim.x + threadIdx.x;
    if (idx >= B_ * N_ * D_) return;
    int b   = idx >> 16;          // / (N*D) == 65536
    int rem = idx & 65535;
    int n = rem >> 6, d = rem & 63;
    u16 zb = f2bf(zsrc[(size_t)b * zstride + rem]);
    u16 hb = f2bf(hsrc[(size_t)b * hstride + rem]);
    zbf[idx] = zb;  hbf[idx] = hb;
    zT[((size_t)b * D_ + d) * N_ + n] = zb;
    hT[((size_t)b * D_ + d) * N_ + n] = hb;
}

// ---------------- q/k projections: [B*N,64] @ [64,64] (x2 weights, x2 paths) ------------------
__global__ __launch_bounds__(32) void proj_kernel(
    const u16* __restrict__ zbf, const u16* __restrict__ hbf,
    const u16* __restrict__ wqx, const u16* __restrict__ wkx,
    const u16* __restrict__ wqh, const u16* __restrict__ wkh,
    u16* __restrict__ qx, u16* __restrict__ kx,
    u16* __restrict__ qh, u16* __restrict__ kh) {
    const int lane = threadIdx.x, l15 = lane & 15, h = lane >> 4;
    const int g0 = blockIdx.x * 16;   // global row over B*N
    const u16 *src, *wq, *wk;  u16 *qo, *ko;
    if (blockIdx.y == 0) { src = zbf; wq = wqx; wk = wkx; qo = qx; ko = kx; }
    else                 { src = hbf; wq = wqh; wk = wkh; qo = qh; ko = kh; }

    v16bf a0 = load_a16(src + (size_t)(g0 + l15) * D_ + 0  + 16 * h);
    v16bf a1 = load_a16(src + (size_t)(g0 + l15) * D_ + 32 + 16 * h);

#pragma unroll
    for (int which = 0; which < 2; ++which) {
        const u16* w = which ? wk : wq;
        u16*       o = which ? ko : qo;
#pragma unroll
        for (int nt = 0; nt < 4; ++nt) {
            v8f c = vzero();
            v16bf b0 = load_b16(w + (nt * 16 + l15) * D_ + 0  + 16 * h);
            v16bf b1 = load_b16(w + (nt * 16 + l15) * D_ + 32 + 16 * h);
            c = wmma_bf16(a0, b0, c);
            c = wmma_bf16(a1, b1, c);
#pragma unroll
            for (int r = 0; r < 8; ++r)
                o[(size_t)(g0 + r + 8 * h) * D_ + nt * 16 + l15] = f2bf(c[r]);
        }
    }
}

// ---------------- graph attention: scores -> softmax (LDS) -> P @ z -------------------------
// grid: (N/16 row tiles, B, 2 paths). 1 wave per block. 32KB LDS holds 16x1024 bf16 scores/probs.
__global__ __launch_bounds__(32) void attend_kernel(
    const u16* __restrict__ qx, const u16* __restrict__ kx,
    const u16* __restrict__ zT, u16* __restrict__ gx,
    const u16* __restrict__ qh, const u16* __restrict__ kh,
    const u16* __restrict__ hT, u16* __restrict__ gh,
    float* __restrict__ attnx, float* __restrict__ attnh) {
    __shared__ u16 sbuf[16 * 1024];
#if USE_ASYNC_LDS
    __shared__ u16 kstage[2][16 * 64];     // double-buffered k tile, 2KB each
#endif
    const int lane = threadIdx.x, l15 = lane & 15, h = lane >> 4;
    const int n0 = blockIdx.x * 16;
    const int b  = blockIdx.y;
    const u16 *q, *k, *vT;  u16* g;  float* attn;
    if (blockIdx.z == 0) { q = qx; k = kx; vT = zT; g = gx; attn = attnx; }
    else                 { q = qh; k = kh; vT = hT; g = gh; attn = attnh; }
    const u16* qb = q  + (size_t)b * N_ * D_;
    const u16* kb = k  + (size_t)b * N_ * D_;
    const u16* vb = vT + (size_t)b * D_ * N_;
    u16*       gb = g  + (size_t)b * N_ * D_;

    v16bf aq0 = load_a16(qb + (size_t)(n0 + l15) * D_ + 0  + 16 * h);
    v16bf aq1 = load_a16(qb + (size_t)(n0 + l15) * D_ + 32 + 16 * h);

    float m8[8];
#pragma unroll
    for (int r = 0; r < 8; ++r) m8[r] = -3.0e38f;

#if USE_ASYNC_LDS
    // Async-copy one 16x64 bf16 k tile (2KB) into LDS: 4 x b128 per wave (16B per lane).
    auto issue_copy = [&](int ct, int buf) {
        const char* src = (const char*)(kb + (size_t)ct * 16 * D_);
        char*       dst = (char*)&kstage[buf][0];
#pragma unroll
        for (int c = 0; c < 4; ++c) {
            int off = c * 512 + lane * 16;
            __builtin_amdgcn_global_load_async_to_lds_b128(
                (AS1 v4i*)(src + off), (AS3 v4i*)(dst + off), 0, 0);
        }
    };
    issue_copy(0, 0);
#endif

    // Phase 1: raw scores (scaled 1/sqrt(D)) into LDS as bf16; per-lane running row-max.
    for (int ct = 0; ct < 64; ++ct) {
        const int m0 = ct * 16;
#if USE_ASYNC_LDS
        if (ct + 1 < 64) { issue_copy(ct + 1, (ct + 1) & 1); __builtin_amdgcn_s_wait_asynccnt(4); }
        else             { __builtin_amdgcn_s_wait_asynccnt(0); }
        const u16* kt0 = &kstage[ct & 1][0];
        v16bf bk0 = load_b16(kt0 + l15 * D_ + 0  + 16 * h);
        v16bf bk1 = load_b16(kt0 + l15 * D_ + 32 + 16 * h);
#else
        if (ct + 4 < 64)
            __builtin_prefetch(kb + (size_t)((ct + 4) * 16 + l15) * D_, 0, 1);
        v16bf bk0 = load_b16(kb + (size_t)(m0 + l15) * D_ + 0  + 16 * h);
        v16bf bk1 = load_b16(kb + (size_t)(m0 + l15) * D_ + 32 + 16 * h);
#endif
        v8f c = vzero();
        c = wmma_bf16(aq0, bk0, c);
        c = wmma_bf16(aq1, bk1, c);
#pragma unroll
        for (int r = 0; r < 8; ++r) {
            float s = c[r] * 0.125f;   // 1/sqrt(64)
            m8[r] = fmaxf(m8[r], s);
            sbuf[(r + 8 * h) * 1024 + m0 + l15] = f2bf(s);
        }
    }
    // Cross-lane (16-wide) max reduction, once per row register.
#pragma unroll
    for (int r = 0; r < 8; ++r) {
        float t = m8[r];
        t = fmaxf(t, __shfl_xor(t, 1, 16));
        t = fmaxf(t, __shfl_xor(t, 2, 16));
        t = fmaxf(t, __shfl_xor(t, 4, 16));
        t = fmaxf(t, __shfl_xor(t, 8, 16));
        m8[r] = t;
    }

    // Phase 2: exponentiate in place, accumulate row sums.
    float sum8[8];
#pragma unroll
    for (int r = 0; r < 8; ++r) sum8[r] = 0.f;
    for (int ct = 0; ct < 64; ++ct) {
#pragma unroll
        for (int r = 0; r < 8; ++r) {
            int idx = (r + 8 * h) * 1024 + ct * 16 + l15;
            float e = __expf(bf2f(sbuf[idx]) - m8[r]);
            sum8[r] += e;
            sbuf[idx] = f2bf(e);
        }
    }
    float rinv[8];
#pragma unroll
    for (int r = 0; r < 8; ++r) {
        float t = sum8[r];
        t += __shfl_xor(t, 1, 16);
        t += __shfl_xor(t, 2, 16);
        t += __shfl_xor(t, 4, 16);
        t += __shfl_xor(t, 8, 16);
        rinv[r] = 1.0f / t;
    }

    // Phase 3: attended = P @ z  (A from LDS, B from transposed z in global).
    v8f acc[4];
#pragma unroll
    for (int nt = 0; nt < 4; ++nt) acc[nt] = vzero();
    for (int kt = 0; kt < 32; ++kt) {
        v16bf ap = load_a16(&sbuf[l15 * 1024 + kt * 32 + 16 * h]);
#pragma unroll
        for (int nt = 0; nt < 4; ++nt) {
            v16bf bv = load_b16(vb + (size_t)(nt * 16 + l15) * N_ + kt * 32 + 16 * h);
            acc[nt] = wmma_bf16(ap, bv, acc[nt]);
        }
    }
#pragma unroll
    for (int nt = 0; nt < 4; ++nt)
#pragma unroll
        for (int r = 0; r < 8; ++r)
            gb[(size_t)(n0 + r + 8 * h) * D_ + nt * 16 + l15] = f2bf(acc[nt][r] * rinv[r]);

    // Last timestep: emit normalized attention matrix (fp32) to d_out.
    if (attn) {
        float* ab = attn + (size_t)b * (size_t)L_ * N_ * N_;
        for (int ct = 0; ct < 64; ++ct) {
#pragma unroll
            for (int r = 0; r < 8; ++r) {
                int row = r + 8 * h;
                float pv = bf2f(sbuf[row * 1024 + ct * 16 + l15]) * rinv[r];
                ab[(size_t)(n0 + row) * N_ + ct * 16 + l15] = pv;
            }
        }
    }
}

// ---------------- gates + GRU update ---------------------------------------------------------
__global__ __launch_bounds__(32) void gates_kernel(
    const u16* __restrict__ gxb, const u16* __restrict__ ghb,
    const u16* __restrict__ wx,  const u16* __restrict__ wh,
    const float* __restrict__ bias,          // b[l][0..191]
    float* __restrict__ hidden,              // [B,L,N,D] (updated in place)
    int l, float* __restrict__ outp, int t) {
    const int lane = threadIdx.x, l15 = lane & 15, h = lane >> 4;
    const int g0 = blockIdx.x * 16;          // global row over B*N
    const int b = g0 >> 10, nloc = g0 & 1023;

    v16bf ax0 = load_a16(gxb + (size_t)(g0 + l15) * D_ + 0  + 16 * h);
    v16bf ax1 = load_a16(gxb + (size_t)(g0 + l15) * D_ + 32 + 16 * h);
    v16bf ah0 = load_a16(ghb + (size_t)(g0 + l15) * D_ + 0  + 16 * h);
    v16bf ah1 = load_a16(ghb + (size_t)(g0 + l15) * D_ + 32 + 16 * h);
    float* hb = hidden + ((size_t)b * L_ + l) * (size_t)N_ * D_;

#pragma unroll
    for (int nt = 0; nt < 4; ++nt) {
        v8f ci[3], ch[3];
#pragma unroll
        for (int p = 0; p < 3; ++p) { ci[p] = vzero(); ch[p] = vzero(); }
#pragma unroll
        for (int p = 0; p < 3; ++p) {
            int jrow = p * 64 + nt * 16 + l15;
            v16bf bx0 = load_b16(wx + (size_t)jrow * D_ + 0  + 16 * h);
            v16bf bx1 = load_b16(wx + (size_t)jrow * D_ + 32 + 16 * h);
            ci[p] = wmma_bf16(ax0, bx0, ci[p]);
            ci[p] = wmma_bf16(ax1, bx1, ci[p]);
            v16bf bh0 = load_b16(wh + (size_t)jrow * D_ + 0  + 16 * h);
            v16bf bh1 = load_b16(wh + (size_t)jrow * D_ + 32 + 16 * h);
            ch[p] = wmma_bf16(ah0, bh0, ch[p]);
            ch[p] = wmma_bf16(ah1, bh1, ch[p]);
        }
        int col = nt * 16 + l15;
        float bz = bias[col], br = bias[col + 64], bn = bias[col + 128];
#pragma unroll
        for (int r = 0; r < 8; ++r) {
            int row = nloc + r + 8 * h;
            float hold = hb[(size_t)row * D_ + col];
            float zg = 1.f / (1.f + __expf(-(ci[0][r] + bz + ch[0][r])));
            float rg = 1.f / (1.f + __expf(-(ci[1][r] + br + ch[1][r])));
            float ng = tanhf(ci[2][r] + bn + rg * ch[2][r]);
            float hnew = (1.f - zg) * ng + zg * hold;
            hb[(size_t)row * D_ + col] = hnew;
            if (outp)
                outp[(((size_t)b * T_ + t) * N_ + row) * D_ + col] = hnew;
        }
    }
}

// ---------------------------------------------------------------------------------------------
extern "C" void kernel_launch(void* const* d_in, const int* in_sizes, int n_in,
                              void* d_out, int out_size, void* d_ws, size_t ws_size,
                              hipStream_t stream) {
    (void)in_sizes; (void)n_in; (void)out_size; (void)ws_size;
    const float* x    = (const float*)d_in[0];
    const float* Wq_x = (const float*)d_in[1];
    const float* Wk_x = (const float*)d_in[2];
    const float* Wq_h = (const float*)d_in[3];
    const float* Wk_h = (const float*)d_in[4];
    const float* Wx   = (const float*)d_in[5];
    const float* Wh   = (const float*)d_in[6];
    const float* bvec = (const float*)d_in[7];

    float* out = (float*)d_out;
    const size_t OUT_ELEMS = (size_t)B_ * T_ * N_ * D_;
    const size_t HID_ELEMS = (size_t)B_ * L_ * N_ * D_;
    const size_t ATT_ELEMS = (size_t)B_ * L_ * N_ * N_;
    float* hidden = out + OUT_ELEMS;
    float* attnx  = hidden + HID_ELEMS;
    float* attnh  = attnx + ATT_ELEMS;

    // workspace carve (all 256B aligned)
    char* p = (char*)d_ws;
    auto alloc = [&](size_t bytes) -> void* {
        void* r = (void*)p;
        p += (bytes + 255) & ~(size_t)255;
        return r;
    };
    u16* wqx = (u16*)alloc((size_t)L_ * D_ * D_ * 2);
    u16* wkx = (u16*)alloc((size_t)L_ * D_ * D_ * 2);
    u16* wqh = (u16*)alloc((size_t)L_ * D_ * D_ * 2);
    u16* wkh = (u16*)alloc((size_t)L_ * D_ * D_ * 2);
    u16* wxw = (u16*)alloc((size_t)L_ * D3_ * D_ * 2);
    u16* whw = (u16*)alloc((size_t)L_ * D3_ * D_ * 2);
    const size_t BND = (size_t)B_ * N_ * D_;
    u16* zbf = (u16*)alloc(BND * 2);
    u16* zT  = (u16*)alloc(BND * 2);
    u16* hbf = (u16*)alloc(BND * 2);
    u16* hT  = (u16*)alloc(BND * 2);
    u16* qx  = (u16*)alloc(BND * 2);
    u16* kx  = (u16*)alloc(BND * 2);
    u16* qh  = (u16*)alloc(BND * 2);
    u16* kh  = (u16*)alloc(BND * 2);
    u16* gx  = (u16*)alloc(BND * 2);
    u16* gh  = (u16*)alloc(BND * 2);

    // h0 = 0 (hidden lives in d_out and is updated in place)
    (void)hipMemsetAsync(hidden, 0, HID_ELEMS * sizeof(float), stream);

    // bf16 transposed weights
    {
        int n1 = L_ * D_ * D_, n2 = L_ * D_ * D3_;
        wconv_kernel<<<(n1 + 255) / 256, 256, 0, stream>>>(Wq_x, wqx, D_, D_);
        wconv_kernel<<<(n1 + 255) / 256, 256, 0, stream>>>(Wk_x, wkx, D_, D_);
        wconv_kernel<<<(n1 + 255) / 256, 256, 0, stream>>>(Wq_h, wqh, D_, D_);
        wconv_kernel<<<(n1 + 255) / 256, 256, 0, stream>>>(Wk_h, wkh, D_, D_);
        wconv_kernel<<<(n2 + 255) / 256, 256, 0, stream>>>(Wx, wxw, D_, D3_);
        wconv_kernel<<<(n2 + 255) / 256, 256, 0, stream>>>(Wh, whw, D_, D3_);
    }

    const int nElems = B_ * N_ * D_;
    for (int t = 0; t < T_; ++t) {
        for (int l = 0; l < L_; ++l) {
            const float* zsrc;  long long zstride;
            if (l == 0) { zsrc = x + (size_t)t * N_ * D_;             zstride = (long long)T_ * N_ * D_; }
            else        { zsrc = hidden + (size_t)(l - 1) * N_ * D_;  zstride = (long long)L_ * N_ * D_; }
            const float* hsrc = hidden + (size_t)l * N_ * D_;
            long long hstride = (long long)L_ * N_ * D_;

            tobf16_kernel<<<(nElems + 255) / 256, 256, 0, stream>>>(
                zsrc, zstride, hsrc, hstride, zbf, zT, hbf, hT);

            proj_kernel<<<dim3(B_ * N_ / 16, 2), 32, 0, stream>>>(
                zbf, hbf,
                wqx + (size_t)l * D_ * D_, wkx + (size_t)l * D_ * D_,
                wqh + (size_t)l * D_ * D_, wkh + (size_t)l * D_ * D_,
                qx, kx, qh, kh);

            bool last = (t == T_ - 1);
            attend_kernel<<<dim3(N_ / 16, B_, 2), 32, 0, stream>>>(
                qx, kx, zT, gx, qh, kh, hT, gh,
                last ? (attnx + (size_t)l * N_ * N_) : (float*)nullptr,
                last ? (attnh + (size_t)l * N_ * N_) : (float*)nullptr);

            gates_kernel<<<B_ * N_ / 16, 32, 0, stream>>>(
                gx, gh,
                wxw + (size_t)l * D3_ * D_, whw + (size_t)l * D3_ * D_,
                bvec + (size_t)l * D3_, hidden, l,
                (l == L_ - 1) ? out : (float*)nullptr, t);
        }
    }
}
